// ConvAttentionCUDA_2302102470965
// MI455X (gfx1250) — matrix operand order repaired and checked by reading, not compiled, not run on hardware
//
#include <hip/hip_runtime.h>
#include <hip/hip_bf16.h>
#include <math.h>

typedef __attribute__((ext_vector_type(16))) _Float16 v16h;
typedef __attribute__((ext_vector_type(8)))  float    v8f;

namespace {
constexpr int Hh   = 16;
constexpr int Sseq = 2048;
constexpr int Dh   = 128;
constexpr int BQ   = 64;
constexpr int BK   = 64;
constexpr int NQB  = Sseq / BQ;      // 32
constexpr int QROWS = BQ + 16;       // 80 (16-row top halo for conv rows q-2,q-1)
// LDS strides: multiples of 8 f16 (16B) for ds_load_b128 fragments, chosen so that
// 16 consecutive rows map to 16 distinct banks (stride/2 mod 64 generates them).
constexpr int QS_STRIDE = 136;       // f16 (68 dwords -> bank stride 4)
constexpr int KS_STRIDE = 136;       // f16
constexpr int VT_STRIDE = 72;        // f16 (36 dwords -> 36*r mod 64 distinct r=0..15)
constexpr int PS_STRIDE = 72;        // f16
constexpr int SS_STRIDE = 65;        // f32 score tile (row-contig conv reads; odd = no col-bank storm)
// strides in uint4 units (8 f16)
constexpr int QS4 = QS_STRIDE / 8;   // 17
constexpr int KS4 = KS_STRIDE / 8;   // 17
constexpr int VT4 = VT_STRIDE / 8;   // 9
constexpr int PS4 = PS_STRIDE / 8;   // 9

constexpr size_t SMEM_BYTES =
    (size_t)QROWS * QS_STRIDE * 2 +          // Qs      21760
    (size_t)BK    * KS_STRIDE * 2 +          // Ks      17408
    (size_t)2 * Dh * VT_STRIDE * 2 +         // Vt x2   36864
    (size_t)2 * QROWS * SS_STRIDE * 4 +      // S  x2   41600
    (size_t)BQ * PS_STRIDE * 2 +             // Ps       9216
    (size_t)QROWS * 4 +                      // left-halo column
    (size_t)3 * BQ * 4;                      // m, l, alpha
} // namespace

union FragU { v16h v; uint4 q[2]; };

// A matrix 16x32 f16 (row per lane m=lane&15): dwords {o..o+3, o+8..o+11}, o=(lane>>4)*4
// -> two 16B loads at uint4 idx row*stride4 + ch*4 + hi and +2
__device__ __forceinline__ v16h load_fragA(const uint4* p, int stride4, int row0,
                                           int ch, int lane) {
  const int idx = (row0 + (lane & 15)) * stride4 + ch * 4 + (lane >> 4);
  FragU f;
  f.q[0] = p[idx];
  f.q[1] = p[idx + 2];
  return f.v;
}

// B matrix 32x16 f16 (column per lane n=lane&15): same K order as A, dwords {o..o+3, o+8..o+11}
// -> two 16B loads at uint4 idx row*stride4 + ch*4 + hi and +2
__device__ __forceinline__ v16h load_fragB(const uint4* p, int stride4, int nrow0,
                                           int ch, int lane) {
  const int idx = (nrow0 + (lane & 15)) * stride4 + ch * 4 + (lane >> 4);
  FragU f;
  f.q[0] = p[idx];
  f.q[1] = p[idx + 2];
  return f.v;
}

// Conv(3x3) + online softmax + rescale-and-PV for one deferred 64-column block.
__device__ __forceinline__ void process_block(
    int q0, int cprev0,
    const float* __restrict__ Sp,   // score tile of deferred block (QROWS x SS_STRIDE)
    const float* __restrict__ Sr,   // right-halo source (col 0 of next block), always valid
    float rmul,                     // 1.0 if Sr real, 0.0 for the last block
    float* halo, float* m_s, float* l_s, float* a_s, _Float16* Ps,
    const _Float16* __restrict__ Vtp,
    const float (&w)[3][3], v8f (&O)[4],
    int tid, int lane, int mtO, int ntO0)
{
  { // ---- 3x3 depthwise conv + online softmax: 4 threads per row, 16 cols each ----
    const int row = tid >> 2;        // 0..63 output row
    const int cg  = tid & 3;         // column group
    const int br  = row + 16;        // buffer row
    const int gq  = q0 + row;
    float rows[3][18];
#pragma unroll
    for (int i = 0; i < 3; ++i) {
      const int rr = br - 2 + i;     // >= 14, top halo rows always in-buffer
      const float* srow = Sp + rr * SS_STRIDE;
      const float lh = halo[rr];
      const float rh = Sr[rr * SS_STRIDE] * rmul;
#pragma unroll
      for (int j = 0; j < 18; ++j) {
        const int col = cg * 16 - 1 + j;
        float v;
        if (col < 0)       v = lh;       // k = cprev0-1 (saved halo / zero pad)
        else if (col > 63) v = rh;       // k = cprev0+64 (next block col 0 / zero)
        else               v = srow[col];
        rows[i][j] = v;
      }
    }
    float pv[16];
    float tmax = -INFINITY;
#pragma unroll
    for (int ci = 0; ci < 16; ++ci) {
      float acc = 0.f;
#pragma unroll
      for (int i = 0; i < 3; ++i)
#pragma unroll
        for (int j = 0; j < 3; ++j)
          acc += w[i][j] * rows[i][ci + j];
      pv[ci] = acc;
      if (cprev0 + cg * 16 + ci <= gq) tmax = fmaxf(tmax, acc);   // causal
    }
    // reduce across the 4 lanes sharing this row (lanes aligned to 4 in-wave, wave32)
    tmax = fmaxf(tmax, __shfl_xor(tmax, 1, 32));
    tmax = fmaxf(tmax, __shfl_xor(tmax, 2, 32));
    const float m_old = m_s[row];
    const float mnew  = fmaxf(m_old, tmax);
    float alpha = 1.f, sum = 0.f;
    if (mnew == -INFINITY) {           // fully masked block for this row
#pragma unroll
      for (int ci = 0; ci < 16; ++ci) Ps[row * PS_STRIDE + cg * 16 + ci] = (_Float16)0.f;
    } else {
      alpha = __expf(m_old - mnew);    // exp(-inf)=0 on first live block
#pragma unroll
      for (int ci = 0; ci < 16; ++ci) {
        const bool ok = (cprev0 + cg * 16 + ci) <= gq;
        const float p = ok ? __expf(pv[ci] - mnew) : 0.f;
        sum += p;
        Ps[row * PS_STRIDE + cg * 16 + ci] = (_Float16)p;
      }
    }
    sum += __shfl_xor(sum, 1, 32);
    sum += __shfl_xor(sum, 2, 32);
    if (cg == 0) { m_s[row] = mnew; l_s[row] = l_s[row] * alpha + sum; a_s[row] = alpha; }
  }
  __syncthreads();
  // save this block's last column as the next block's left halo (halo reads done above)
  if (tid < QROWS) halo[tid] = Sp[tid * SS_STRIDE + (BK - 1)];
  { // ---- rescale O by alpha and accumulate P' x V via WMMA ----
    const int hi = lane >> 4;
    float ar[8];
#pragma unroll
    for (int r = 0; r < 8; ++r) ar[r] = a_s[mtO * 16 + r + 8 * hi];
#pragma unroll
    for (int t = 0; t < 4; ++t)
#pragma unroll
      for (int r = 0; r < 8; ++r) O[t][r] *= ar[r];
    const uint4* Pp = (const uint4*)Ps;
    const uint4* Vp = (const uint4*)Vtp;
#pragma unroll
    for (int t = 0; t < 4; ++t) {
      v8f c = O[t];
#pragma unroll
      for (int ch = 0; ch < 2; ++ch) {
        v16h a = load_fragA(Pp, PS4, mtO * 16, ch, lane);
        v16h b = load_fragB(Vp, VT4, (ntO0 + t) * 16, ch, lane);
        c = __builtin_amdgcn_wmma_f32_16x16x32_f16(false, a, false, b, (short)0, c,
                                                   false, false);
      }
      O[t] = c;
    }
  }
  __syncthreads();
}

__global__ __launch_bounds__(256)
void conv_attn_kernel(const float* __restrict__ Qg, const float* __restrict__ Kg,
                      const float* __restrict__ Vg, const float* __restrict__ Wg,
                      float* __restrict__ Og) {
  extern __shared__ __align__(16) char smem[];
  _Float16* Qs   = (_Float16*)smem;                               // QROWS x QS_STRIDE
  _Float16* Ks   = Qs + QROWS * QS_STRIDE;                        // BK x KS_STRIDE
  _Float16* Vt   = Ks + BK * KS_STRIDE;                           // 2 x Dh x VT_STRIDE
  float*    Ssb  = (float*)(Vt + 2 * Dh * VT_STRIDE);             // 2 x QROWS x SS_STRIDE
  _Float16* Ps   = (_Float16*)(Ssb + 2 * QROWS * SS_STRIDE);      // BQ x PS_STRIDE
  float*    halo = (float*)(Ps + BQ * PS_STRIDE);                 // QROWS
  float*    m_s  = halo + QROWS;
  float*    l_s  = m_s + BQ;
  float*    a_s  = l_s + BQ;

  const int tid  = threadIdx.x;
  const int wave = tid >> 5, lane = tid & 31;
  const int bid  = blockIdx.x;
  const int h    = bid >> 5;          // NQB == 32
  const int qb   = bid & 31;
  const int q0   = qb * BQ;
  const float scale = 0.08838834764831845f;   // 1/sqrt(128)

  float w[3][3];
#pragma unroll
  for (int i = 0; i < 3; ++i)
#pragma unroll
    for (int j = 0; j < 3; ++j) w[i][j] = Wg[h * 9 + i * 3 + j];

  const float* Qh = Qg + (size_t)h * Sseq * Dh;
  const float* Kh = Kg + (size_t)h * Sseq * Dh;
  const float* Vh = Vg + (size_t)h * Sseq * Dh;

  // Q tile: rows q0-16 .. q0+63, float4 global loads -> packed f16 dwords
  for (int i = tid; i < QROWS * (Dh / 4); i += 256) {
    const int r  = i >> 5;            // Dh/4 == 32
    const int c4 = (i & 31) * 4;
    int gq = q0 - 16 + r; if (gq < 0) gq = 0;   // clamped rows are causally zeroed later
    const float4 qv = *(const float4*)(Qh + (size_t)gq * Dh + c4);
    union { _Float16 hx[4]; unsigned u[2]; } pk;
    pk.hx[0] = (_Float16)qv.x; pk.hx[1] = (_Float16)qv.y;
    pk.hx[2] = (_Float16)qv.z; pk.hx[3] = (_Float16)qv.w;
    unsigned* qd = (unsigned*)(Qs + r * QS_STRIDE + c4);
    qd[0] = pk.u[0]; qd[1] = pk.u[1];
  }
  for (int i = tid; i < BQ; i += 256) { m_s[i] = -INFINITY; l_s[i] = 0.f; a_s[i] = 1.f; }
  for (int i = tid; i < QROWS; i += 256) halo[i] = 0.f;

  // Output accumulators: wave owns M-tile (wave>>1), N-tiles (wave&1)*4 .. +3
  const int mtO  = wave >> 1;
  const int ntO0 = (wave & 1) * 4;
  v8f O[4];
#pragma unroll
  for (int t = 0; t < 4; ++t) O[t] = {};

  const int NB = qb + 1;               // k blocks covering 0..q0+BQ-1
  for (int kb = 0; kb < NB; ++kb) {
    const int c0 = kb * BK;
    const int pb = kb & 1;
    // stage K (row-major f16) and V (transposed d-major f16); float4 global loads
    for (int i = tid; i < BK * (Dh / 4); i += 256) {
      const int r  = i >> 5;
      const int c4 = (i & 31) * 4;
      const float4 kv = *(const float4*)(Kh + (size_t)(c0 + r) * Dh + c4);
      const float4 vv = *(const float4*)(Vh + (size_t)(c0 + r) * Dh + c4);
      union { _Float16 hx[4]; unsigned u[2]; } pk;
      pk.hx[0] = (_Float16)kv.x; pk.hx[1] = (_Float16)kv.y;
      pk.hx[2] = (_Float16)kv.z; pk.hx[3] = (_Float16)kv.w;
      unsigned* kd = (unsigned*)(Ks + r * KS_STRIDE + c4);
      kd[0] = pk.u[0]; kd[1] = pk.u[1];
      _Float16* vt = Vt + pb * Dh * VT_STRIDE + r;
      vt[(size_t)(c4 + 0) * VT_STRIDE] = (_Float16)vv.x;
      vt[(size_t)(c4 + 1) * VT_STRIDE] = (_Float16)vv.y;
      vt[(size_t)(c4 + 2) * VT_STRIDE] = (_Float16)vv.z;
      vt[(size_t)(c4 + 3) * VT_STRIDE] = (_Float16)vv.w;
    }
    if (kb + 1 < NB) {   // prefetch next K/V block (global_prefetch_b8)
      const float* nk = Kh + (size_t)(c0 + BK) * Dh;
      const float* nv = Vh + (size_t)(c0 + BK) * Dh;
      __builtin_prefetch(nk + (size_t)tid * 32, 0, 1);
      __builtin_prefetch(nv + (size_t)tid * 32, 0, 1);
    }
    __syncthreads();
    // Q x K^T -> S tile (causally zeroed, scaled). 20 tiles over 8 waves.
    float* Sb = Ssb + pb * QROWS * SS_STRIDE;
    const uint4* Qp = (const uint4*)Qs;
    const uint4* Kp = (const uint4*)Ks;
    for (int t = wave; t < 20; t += 8) {
      const int mt = t >> 2, nt = t & 3;
      v8f c = {};
#pragma unroll
      for (int ch = 0; ch < 4; ++ch) {
        v16h a = load_fragA(Qp, QS4, mt * 16, ch, lane);
        v16h b = load_fragB(Kp, KS4, nt * 16, ch, lane);
        c = __builtin_amdgcn_wmma_f32_16x16x32_f16(false, a, false, b, (short)0, c,
                                                   false, false);
      }
      const int rl = lane & 15, hi = lane >> 4;
      const int gk = c0 + nt * 16 + rl;
#pragma unroll
      for (int r = 0; r < 8; ++r) {
        const int row = mt * 16 + r + 8 * hi;
        const int gq  = q0 - 16 + row;
        Sb[row * SS_STRIDE + nt * 16 + rl] = (gk <= gq) ? c[r] * scale : 0.f;
      }
    }
    __syncthreads();
    if (kb > 0) {   // deferred conv+softmax+PV for block kb-1 (right halo = cur col 0)
      process_block(q0, (kb - 1) * BK,
                    Ssb + ((kb - 1) & 1) * QROWS * SS_STRIDE,
                    Ssb + pb * QROWS * SS_STRIDE, 1.f,
                    halo, m_s, l_s, a_s, Ps,
                    Vt + ((kb - 1) & 1) * Dh * VT_STRIDE,
                    w, O, tid, lane, mtO, ntO0);
    }
  }
  { // final block: right halo is provably zero (column NB*BK > every q in tile)
    const float* Sp = Ssb + ((NB - 1) & 1) * QROWS * SS_STRIDE;
    process_block(q0, (NB - 1) * BK, Sp, Sp, 0.f,
                  halo, m_s, l_s, a_s, Ps,
                  Vt + ((NB - 1) & 1) * Dh * VT_STRIDE,
                  w, O, tid, lane, mtO, ntO0);
  }
  // epilogue: O / l -> global (coalesced b32 stores: 16 consecutive d per half-wave)
  {
    const int rl = lane & 15, hi = lane >> 4;
#pragma unroll
    for (int t = 0; t < 4; ++t) {
      const int nt = ntO0 + t;
#pragma unroll
      for (int r = 0; r < 8; ++r) {
        const int row = mtO * 16 + r + 8 * hi;
        const float inv = 1.f / l_s[row];
        Og[((size_t)h * Sseq + (q0 + row)) * Dh + nt * 16 + rl] = O[t][r] * inv;
      }
    }
  }
}

extern "C" void kernel_launch(void* const* d_in, const int* in_sizes, int n_in,
                              void* d_out, int out_size, void* d_ws, size_t ws_size,
                              hipStream_t stream) {
  (void)in_sizes; (void)n_in; (void)out_size; (void)d_ws; (void)ws_size;
  const float* Q = (const float*)d_in[0];
  const float* K = (const float*)d_in[1];
  const float* V = (const float*)d_in[2];
  const float* W = (const float*)d_in[3];
  float* out = (float*)d_out;
  static_assert(SMEM_BYTES < 320 * 1024, "LDS budget");
  (void)hipFuncSetAttribute((const void*)conv_attn_kernel,
                            hipFuncAttributeMaxDynamicSharedMemorySize,
                            (int)SMEM_BYTES);
  dim3 grid(Hh * NQB);
  conv_attn_kernel<<<grid, 256, SMEM_BYTES, stream>>>(Q, K, V, W, out);
}
